// SelfAttention_7713761264304
// MI455X (gfx1250) — matrix operand order repaired and checked by reading, not compiled, not run on hardware
//
#include <hip/hip_runtime.h>
#include <hip/hip_bf16.h>
#include <math.h>

typedef __attribute__((ext_vector_type(16))) _Float16 v16h;
typedef __attribute__((ext_vector_type(8)))  _Float16 v8h;
typedef __attribute__((ext_vector_type(8)))  float    v8f;
typedef int v4i __attribute__((vector_size(16)));

#define B_    2
#define T_    2048
#define LDIM  1024
#define CDIM  1024
#define H_    16
#define HS_   64
#define M_    (B_ * T_)      // 4096 rows total
#define BK    32             // GEMM K-step (one WMMA K)

// ---------------------------------------------------------------------------
// WMMA helpers (CDNA5 wave32, V_WMMA_F32_16X16X32_F16)
// ---------------------------------------------------------------------------
static __device__ __forceinline__ v8f wmma_f16(v16h a, v16h b, v8f c) {
  return __builtin_amdgcn_wmma_f32_16x16x32_f16(
      /*neg_a=*/false, a, /*neg_b=*/false, b,
      /*c_mod=*/(short)0, c, /*reuse_a=*/false, /*reuse_b=*/false);
}

// A-matrix 16x32 f16 fragment (ISA 7.12.2): lane<16 holds row=lane, K{0-7,16-23};
// lane>=16 holds row=lane-16, K{8-15,24-31}. Two 16B chunks per lane.
static __device__ __forceinline__ v16h load_a_frag(const _Float16* base, int ld) {
  const int lane = threadIdx.x & 31;
  const int row  = lane & 15;
  const int koff = (lane >> 4) << 3;          // 0 or 8 halves
  const _Float16* p = base + (size_t)row * ld + koff;
  union { v16h v; v8h h[2]; } u;
  u.h[0] = *reinterpret_cast<const v8h*>(p);
  u.h[1] = *reinterpret_cast<const v8h*>(p + 16);
  return u.v;
}

// B-matrix 32x16 f16 fragment, sourced from B-transposed [N][K] storage:
// lane<16 holds col=lane, K{0-7,16-23}; lane>=16 holds col, K{8-15,24-31}.
static __device__ __forceinline__ v16h load_b_frag(const _Float16* base, int ld) {
  const int lane = threadIdx.x & 31;
  const int col  = lane & 15;
  const int koff = (lane >> 4) << 3;          // 0 or 8 halves
  const _Float16* p = base + (size_t)col * ld + koff;
  union { v16h v; v8h h[2]; } u;
  u.h[0] = *reinterpret_cast<const v8h*>(p);
  u.h[1] = *reinterpret_cast<const v8h*>(p + 16);
  return u.v;
}

// ---------------------------------------------------------------------------
// Async global->LDS copy (CDNA5 GLOBAL_LOAD_ASYNC_TO_LDS_B128, ASYNCcnt).
// Builtin signature (from hipcc diagnostic): (v4i AS1* src, v4i AS3* dst,
// imm offset, imm cpol). Guarded fallback to a synchronous copy.
// ---------------------------------------------------------------------------
#if defined(__has_builtin)
#if __has_builtin(__builtin_amdgcn_global_load_async_to_lds_b128)
#define ATHENA_HAS_ASYNC_LDS 1
#endif
#if __has_builtin(__builtin_amdgcn_s_wait_asynccnt)
#define ATHENA_HAS_ASYNCCNT_BUILTIN 1
#endif
#endif

typedef __attribute__((address_space(1))) v4i* as1_v4i_ptr;
typedef __attribute__((address_space(3))) v4i* as3_v4i_ptr;

static __device__ __forceinline__ void async_copy16(const _Float16* g, _Float16* l) {
#if defined(ATHENA_HAS_ASYNC_LDS)
  // generic->AS1: flat global address is the global address (64-bit);
  // generic->AS3: LDS offset lives in the low 32 bits of the flat address.
  __builtin_amdgcn_global_load_async_to_lds_b128(
      (as1_v4i_ptr)(uintptr_t)g,
      (as3_v4i_ptr)(unsigned int)(uintptr_t)l,
      /*offset=*/0, /*cpol=*/0);
#else
  *reinterpret_cast<v8h*>(l) = *reinterpret_cast<const v8h*>(g);
#endif
}

static __device__ __forceinline__ void async_wait_all() {
#if defined(ATHENA_HAS_ASYNC_LDS)
#if defined(ATHENA_HAS_ASYNCCNT_BUILTIN)
  __builtin_amdgcn_s_wait_asynccnt(0);
#else
  asm volatile("s_wait_asynccnt 0x0" ::: "memory");
#endif
#endif
}

// ---------------------------------------------------------------------------
// Conversions / layout kernels
// ---------------------------------------------------------------------------
__global__ void cvt_f32_to_f16(const float* __restrict__ src,
                               _Float16* __restrict__ dst, int n) {
  for (int i = blockIdx.x * blockDim.x + threadIdx.x; i < n;
       i += gridDim.x * blockDim.x)
    dst[i] = (_Float16)src[i];
}

// wT[n*K + k] = (f16) w[k*N + n]   (weights stored [K][N] f32 -> [N][K] f16)
__global__ void transpose_cvt(const float* __restrict__ w,
                              _Float16* __restrict__ wT, int K, int N) {
  int total = K * N;
  for (int i = blockIdx.x * blockDim.x + threadIdx.x; i < total;
       i += gridDim.x * blockDim.x) {
    int n = i / K;
    int k = i - n * K;
    wT[i] = (_Float16)w[(size_t)k * N + n];
  }
}

// vT[((b*H+h)*64 + d)*T + t] = qkv16[(b*T+t)*3C + 2C + h*64 + d]
__global__ void extract_vT(const _Float16* __restrict__ qkv,
                           _Float16* __restrict__ vT) {
  const int total = B_ * H_ * HS_ * T_;
  for (int i = blockIdx.x * blockDim.x + threadIdx.x; i < total;
       i += gridDim.x * blockDim.x) {
    int t  = i & (T_ - 1);
    int d  = (i >> 11) & (HS_ - 1);
    int bh = i >> 17;
    int b  = bh >> 4;
    int h  = bh & 15;
    vT[i] = qkv[(size_t)(b * T_ + t) * (3 * CDIM) + 2 * CDIM + h * HS_ + d];
  }
}

// ---------------------------------------------------------------------------
// GEMM: C[M,N] = A[M,K](f16) * Wt[N,K](f16)^T + bias[N]
// Block tile 128x128, 4 waves (2x2), wave tile 64x64 (16 WMMAs per K-step).
// A/B K-slabs (128x32 each) are double-buffered in LDS and filled with
// async global->LDS copies; the next slab's copies are kicked off before
// computing the current one (ASYNCcnt + barrier synchronization).
// ---------------------------------------------------------------------------
template <bool F32OUT>
__global__ void __launch_bounds__(128)
gemm_f16(const _Float16* __restrict__ A, const _Float16* __restrict__ Wt,
         const float* __restrict__ bias, void* __restrict__ out,
         int M, int N, int K) {
  __shared__ __align__(16) _Float16 tA[2][128 * BK];   // 8KB per buffer
  __shared__ __align__(16) _Float16 tB[2][128 * BK];

  const int tid  = threadIdx.x;
  const int lane = tid & 31;
  const int wid  = tid >> 5;
  const int wm = wid >> 1, wn = wid & 1;
  const int rowBlk = blockIdx.y * 128;
  const int colBlk = blockIdx.x * 128;

  // Stage mapping: thread t copies row t of the A-slab and row t of the B-slab
  // (64B each = 4 async b128 transfers per slab per thread).
  const _Float16* gA = A  + (size_t)(rowBlk + tid) * K;
  const _Float16* gB = Wt + (size_t)(colBlk + tid) * K;

  v8f acc[4][4];
#pragma unroll
  for (int i = 0; i < 4; ++i)
#pragma unroll
    for (int j = 0; j < 4; ++j) acc[i][j] = {};

  // prologue: stage slab for k0 = 0 into buffer 0
#pragma unroll
  for (int c = 0; c < 4; ++c) {
    async_copy16(gA + 8 * c, &tA[0][tid * BK + 8 * c]);
    async_copy16(gB + 8 * c, &tB[0][tid * BK + 8 * c]);
  }

  int cur = 0;
  for (int k0 = 0; k0 < K; k0 += BK) {
    async_wait_all();          // my slab transfers (and last iter's) arrived
    __syncthreads();           // everyone's arrived; everyone done reading old

    if (k0 + BK < K) {         // kick next slab into the other buffer
      const int nb = cur ^ 1;
#pragma unroll
      for (int c = 0; c < 4; ++c) {
        async_copy16(gA + k0 + BK + 8 * c, &tA[nb][tid * BK + 8 * c]);
        async_copy16(gB + k0 + BK + 8 * c, &tB[nb][tid * BK + 8 * c]);
      }
    }

    // compute on current slab from LDS
    const _Float16* sA = &tA[cur][(wm * 64) * BK];
    const _Float16* sB = &tB[cur][(wn * 64) * BK];
    v16h af[4];
    af[0] = load_a_frag(sA +  0 * BK, BK);
    af[1] = load_a_frag(sA + 16 * BK, BK);
    af[2] = load_a_frag(sA + 32 * BK, BK);
    af[3] = load_a_frag(sA + 48 * BK, BK);

    v16h bf;
#pragma unroll
    for (int j = 0; j < 4; ++j) {
      bf = load_b_frag(sB + 16 * j * BK, BK);
#pragma unroll
      for (int i = 0; i < 4; ++i) acc[i][j] = wmma_f16(af[i], bf, acc[i][j]);
    }
    cur ^= 1;
  }

  const int rb = (lane >> 4) * 8, cc = lane & 15;
  const int row0 = rowBlk + wm * 64;
  const int col0 = colBlk + wn * 64;
#pragma unroll
  for (int i = 0; i < 4; ++i) {
#pragma unroll
    for (int r = 0; r < 8; ++r) {
      const int row = row0 + 16 * i + rb + r;
#pragma unroll
      for (int j = 0; j < 4; ++j) {
        const int col = col0 + 16 * j + cc;
        const float v = acc[i][j][r] + bias[col];
        if (F32OUT) {
          ((float*)out)[(size_t)row * N + col] = v;
        } else {
          ((_Float16*)out)[(size_t)row * N + col] = (_Float16)v;
        }
      }
    }
  }
}

// ---------------------------------------------------------------------------
// Flash attention: one wave per 16-query tile, 64 keys per iteration.
// S = q.k^T via WMMA; online softmax with half-wave shuffles for the row max;
// row SUM via an extra WMMA against an all-ones B operand (same rescaling
// recurrence as O; every column equals the row sum -> no shuffles/broadcast).
// P is re-laid out through LDS into A-fragments; O += P.V with V^T B-frags.
// ---------------------------------------------------------------------------
__global__ void __launch_bounds__(128)
attn_flash(const _Float16* __restrict__ qkv, const _Float16* __restrict__ vT,
           _Float16* __restrict__ ao) {
  __shared__ __align__(16) _Float16 pbuf[4][16 * 64];

  const int wave = threadIdx.x >> 5;
  const int lane = threadIdx.x & 31;
  const int bh = blockIdx.x;
  const int b = bh >> 4, h = bh & 15;
  const int qt = blockIdx.y * 4 + wave;                 // 16-query tile index
  const size_t rowq = (size_t)(b * T_ + qt * 16);
  const int ldq = 3 * CDIM;

  const _Float16* qbase = qkv + rowq * ldq + h * HS_;
  const _Float16* kbase = qkv + (size_t)b * T_ * ldq + CDIM + h * HS_;
  const _Float16* vtb   = vT  + (size_t)(b * H_ + h) * HS_ * T_;

  const v16h qf0 = load_a_frag(qbase + 0,  ldq);        // hs 0..31
  const v16h qf1 = load_a_frag(qbase + 32, ldq);        // hs 32..63

  v16h ones;
#pragma unroll
  for (int t = 0; t < 16; ++t) ones[t] = (_Float16)1.0f;

  v8f o[5];                                             // o[4] = running rowsum
#pragma unroll
  for (int a = 0; a < 5; ++a) o[a] = {};
  float mrow[8];
#pragma unroll
  for (int r = 0; r < 8; ++r) mrow[r] = -INFINITY;

  const float scale = 0.125f;                           // 1/sqrt(64)
  const int rb = (lane >> 4) * 8, cc = lane & 15;
  _Float16* pb = &pbuf[wave][0];

  for (int j = 0; j < T_; j += 64) {
    // ---- scores for 64 keys: four 16x16 C-tiles (8 WMMAs)
    v8f s[4];
#pragma unroll
    for (int t = 0; t < 4; ++t) {
      s[t] = {};
      s[t] = wmma_f16(qf0, load_b_frag(kbase + (size_t)(j + 16 * t) * ldq + 0,  ldq), s[t]);
      s[t] = wmma_f16(qf1, load_b_frag(kbase + (size_t)(j + 16 * t) * ldq + 32, ldq), s[t]);
    }

    // ---- online softmax: row max via 4 half-wave shuffles, exp, rescale
#pragma unroll
    for (int r = 0; r < 8; ++r) {
      float a0 = s[0][r] * scale, a1 = s[1][r] * scale;
      float a2 = s[2][r] * scale, a3 = s[3][r] * scale;
      float rm = fmaxf(fmaxf(a0, a1), fmaxf(a2, a3));
      rm = fmaxf(rm, __shfl_xor(rm, 1));
      rm = fmaxf(rm, __shfl_xor(rm, 2));
      rm = fmaxf(rm, __shfl_xor(rm, 4));
      rm = fmaxf(rm, __shfl_xor(rm, 8));
      const float mn   = fmaxf(mrow[r], rm);
      const float corr = __expf(mrow[r] - mn);
      mrow[r] = mn;
      o[0][r] *= corr; o[1][r] *= corr; o[2][r] *= corr;
      o[3][r] *= corr; o[4][r] *= corr;
      _Float16* prow = pb + (rb + r) * 64 + cc;
      prow[0]  = (_Float16)__expf(a0 - mn);
      prow[16] = (_Float16)__expf(a1 - mn);
      prow[32] = (_Float16)__expf(a2 - mn);
      prow[48] = (_Float16)__expf(a3 - mn);
    }

    // ---- P (16x64) as two A-fragments via per-wave LDS staging
    const v16h pf0 = load_a_frag(pb +  0, 64);          // keys j   .. j+31
    const v16h pf1 = load_a_frag(pb + 32, 64);          // keys j+32.. j+63

    // ---- O += P.V (8 WMMAs) and rowsum += P.1 (2 WMMAs)
    v16h vf;
#pragma unroll
    for (int a = 0; a < 4; ++a) {
      vf = load_b_frag(vtb + (size_t)(16 * a) * T_ + j, T_);
      o[a] = wmma_f16(pf0, vf, o[a]);
    }
    o[4] = wmma_f16(pf0, ones, o[4]);
#pragma unroll
    for (int a = 0; a < 4; ++a) {
      vf = load_b_frag(vtb + (size_t)(16 * a) * T_ + j + 32, T_);
      o[a] = wmma_f16(pf1, vf, o[a]);
    }
    o[4] = wmma_f16(pf1, ones, o[4]);

    if (j + 64 < T_) {                                  // stream-ahead hints
      __builtin_prefetch(kbase + (size_t)(j + 64) * ldq, 0, 1);
      __builtin_prefetch(vtb + (size_t)j + 64, 0, 1);
    }
  }

  // ---- finalize: divide by rowsum (replicated in every column of o[4])
  _Float16* aorow = ao + rowq * CDIM + h * HS_;
#pragma unroll
  for (int r = 0; r < 8; ++r) {
    const float inv = 1.0f / o[4][r];
    const size_t ro = (size_t)(rb + r) * CDIM;
    aorow[ro + cc +  0] = (_Float16)(o[0][r] * inv);
    aorow[ro + cc + 16] = (_Float16)(o[1][r] * inv);
    aorow[ro + cc + 32] = (_Float16)(o[2][r] * inv);
    aorow[ro + cc + 48] = (_Float16)(o[3][r] * inv);
  }
}

// ---------------------------------------------------------------------------
extern "C" void kernel_launch(void* const* d_in, const int* in_sizes, int n_in,
                              void* d_out, int out_size, void* d_ws,
                              size_t ws_size, hipStream_t stream) {
  (void)in_sizes; (void)n_in; (void)out_size; (void)ws_size;

  const float* x      = (const float*)d_in[0];
  const float* w_attn = (const float*)d_in[1];
  const float* b_attn = (const float*)d_in[2];
  const float* w_qkv  = (const float*)d_in[3];
  const float* b_qkv  = (const float*)d_in[4];
  const float* w_proj = (const float*)d_in[5];
  const float* b_proj = (const float*)d_in[6];

  char* ws = (char*)d_ws;
  _Float16* x16   = (_Float16*)ws; ws += (size_t)M_ * LDIM * 2;
  _Float16* waT   = (_Float16*)ws; ws += (size_t)LDIM * CDIM * 2;
  _Float16* wqT   = (_Float16*)ws; ws += (size_t)CDIM * 3 * CDIM * 2;
  _Float16* wpT   = (_Float16*)ws; ws += (size_t)CDIM * CDIM * 2;
  _Float16* h16   = (_Float16*)ws; ws += (size_t)M_ * CDIM * 2;
  _Float16* qkv16 = (_Float16*)ws; ws += (size_t)M_ * 3 * CDIM * 2;
  _Float16* vT16  = (_Float16*)ws; ws += (size_t)B_ * H_ * HS_ * T_ * 2;
  _Float16* ao16  = (_Float16*)ws;

  // f32 -> f16 conversions / weight transposes
  cvt_f32_to_f16<<<1024, 256, 0, stream>>>(x, x16, M_ * LDIM);
  transpose_cvt<<<1024, 256, 0, stream>>>(w_attn, waT, LDIM, CDIM);
  transpose_cvt<<<1024, 256, 0, stream>>>(w_qkv,  wqT, CDIM, 3 * CDIM);
  transpose_cvt<<<1024, 256, 0, stream>>>(w_proj, wpT, CDIM, CDIM);

  // h = x @ W_attn + b    (4096x1024x1024)
  gemm_f16<false><<<dim3(CDIM / 128, M_ / 128), 128, 0, stream>>>(
      x16, waT, b_attn, h16, M_, CDIM, LDIM);

  // qkv = h @ W_qkv + b   (4096x3072x1024)
  gemm_f16<false><<<dim3(3 * CDIM / 128, M_ / 128), 128, 0, stream>>>(
      h16, wqT, b_qkv, qkv16, M_, 3 * CDIM, CDIM);

  // V^T per (b, head) for the P.V WMMA B-operand
  extract_vT<<<1024, 256, 0, stream>>>(qkv16, vT16);

  // flash attention: grid (B*H, T/64), 4 waves/block, 16 queries per wave
  attn_flash<<<dim3(B_ * H_, T_ / 64), 128, 0, stream>>>(qkv16, vT16, ao16);

  // out = ao @ W_proj + b  -> f32 d_out
  gemm_f16<true><<<dim3(CDIM / 128, M_ / 128), 128, 0, stream>>>(
      ao16, wpT, b_proj, d_out, M_, CDIM, CDIM);
}
